// CGCNNB_58815282151652
// MI455X (gfx1250) — matrix operand order repaired and checked by reading, not compiled, not run on hardware
//
#include <hip/hip_runtime.h>
#include <hip/hip_bf16.h>
#include <math.h>

#define N_NODES   50000
#define N_EDGES   800000
#define NUM_GRAPHS 256
#define HD  128
#define CD  129            // CGConv channels
#define ZD  259            // concat dim
#define NLAYERS 5
#define NFC 3

#define KPAD 160           // K padded: 129 -> 160  (5 WMMA k-steps of 32)
#define KSTEPS 5
#define NOUT 516           // 4*129 fused projection outputs
#define NPAD 528           // padded to 33 tiles of 16
#define NT   33            // output column tiles

typedef __attribute__((ext_vector_type(16))) __bf16 v16bf;
typedef __attribute__((ext_vector_type(8)))  __bf16 v8bf;
typedef __attribute__((ext_vector_type(8)))  float  v8f;

// ---------------- setup kernels ----------------

__global__ void k_init_x(const int* __restrict__ atoms, const float* __restrict__ pos,
                         const float* __restrict__ emb, float* __restrict__ x) {
    int i = blockIdx.x * blockDim.x + threadIdx.x;
    if (i >= N_NODES * CD) return;
    int node = i / CD, c = i % CD;
    x[i] = (c < HD) ? emb[atoms[node] * HD + c] : pos[node * 3 + 2];
}

__global__ void k_edge_attr(const int* __restrict__ src, const int* __restrict__ dst,
                            const float* __restrict__ pos, float* __restrict__ ea) {
    int e = blockIdx.x * blockDim.x + threadIdx.x;
    if (e >= N_EDGES) return;
    int s = src[e], d = dst[e];
    float dx = pos[s*3+0] - pos[d*3+0];
    float dy = pos[s*3+1] - pos[d*3+1];
    float dz = pos[s*3+2] - pos[d*3+2];
    ea[e] = sqrtf(dx*dx + dy*dy + dz*dz);
}

// fp32 x [N,129] -> bf16 xb [N,160], zero-padded K so the WMMA k-loop is branch-free
__global__ void k_cvt_x(const float* __restrict__ x, __bf16* __restrict__ xb) {
    int i = blockIdx.x * blockDim.x + threadIdx.x;
    if (i >= N_NODES * KPAD) return;
    int node = i / KPAD, c = i % KPAD;
    xb[i] = (c < CD) ? (__bf16)x[node * CD + c] : (__bf16)0.0f;
}

// Build WT[n][k] (bf16, transposed) fusing 4 projections:
//   n in [0,129)   : Wf[k][n]          (x_dst, filter)
//   n in [129,258) : Wf[129+k][n-129]  (x_src, filter)
//   n in [258,387) : Ws[k][n-258]      (x_dst, soft)
//   n in [387,516) : Ws[129+k][n-387]  (x_src, soft)
// Transposed layout => each WMMA B-operand lane reads 32 contiguous bytes.
__global__ void k_prep_w(const float* __restrict__ Wf, const float* __restrict__ Ws,
                         __bf16* __restrict__ WT) {
    int i = blockIdx.x * blockDim.x + threadIdx.x;
    if (i >= NPAD * KPAD) return;
    int n = i / KPAD, k = i % KPAD;
    float v = 0.0f;
    if (k < CD) {
        if      (n < CD)     v = Wf[k * CD + n];
        else if (n < 2*CD)   v = Wf[(CD + k) * CD + (n - CD)];
        else if (n < 3*CD)   v = Ws[k * CD + (n - 2*CD)];
        else if (n < NOUT)   v = Ws[(CD + k) * CD + (n - 3*CD)];
    }
    WT[n * KPAD + k] = (__bf16)v;
}

// ---------------- node GEMM: Y[N,528] = xb[N,160] @ W[160,528] via bf16 WMMA ----------------
// One block = one 16-row m-tile (3125 blocks). 8 waves stride the 33 n-tiles.
// Each wave preloads ALL 5 A k-step operands into registers once and reuses them
// across its 4-5 n-tiles: 20-25 WMMAs/wave, ~4x less A-operand traffic.
__global__ void __launch_bounds__(256) k_node_gemm(const __bf16* __restrict__ xb,
                                                   const __bf16* __restrict__ WT,
                                                   float* __restrict__ Y) {
    const int lane = threadIdx.x & 31;
    const int wave = threadIdx.x >> 5;
    const int m0 = blockIdx.x * 16;            // 3125*16 == 50000 exactly

    // A layout (16-bit A 16x32): lane&15 = row; lanes16-31 K-offset +16;
    //   elems 0..15 = K base+0..15 (contiguous, matches B)
    const int arow  = lane & 15;
    const int ahalf = (lane >> 4) ? 16 : 0;
    // B layout (32x16): lane&15 = col; lanes16-31 hold K 16..31; K ascending 2/VGPR
    const int bhalf = (lane >> 4) ? 16 : 0;

    const __bf16* ap = xb + (size_t)(m0 + arow) * KPAD + ahalf;

    v16bf A[KSTEPS];
    #pragma unroll
    for (int kk = 0; kk < KSTEPS; ++kk) {
        v8bf a_lo = *(const v8bf*)(ap + kk * 32);
        v8bf a_hi = *(const v8bf*)(ap + kk * 32 + 8);
        A[kk] = __builtin_shufflevector(a_lo, a_hi,
                    0,1,2,3,4,5,6,7,8,9,10,11,12,13,14,15);
    }

    #pragma unroll
    for (int t = 0; t < 5; ++t) {
        const int nt = wave + t * 8;           // wave-uniform; EXEC all-1 at WMMA
        if (nt < NT) {
            const __bf16* bp = WT + (size_t)(nt * 16 + (lane & 15)) * KPAD + bhalf;
            v8f acc = {};
            #pragma unroll
            for (int kk = 0; kk < KSTEPS; ++kk) {
                v16bf B = *(const v16bf*)(bp + kk * 32);
                acc = __builtin_amdgcn_wmma_f32_16x16x32_bf16(
                          false, A[kk], false, B, (short)0, acc, false, false);
            }
            // C/D layout: VGPR r -> row r (lanes 0-15) / row r+8 (lanes 16-31)
            float* yp = Y + (size_t)(m0 + ((lane >> 4) ? 8 : 0)) * NPAD
                          + nt * 16 + (lane & 15);
            #pragma unroll
            for (int r = 0; r < 8; ++r) yp[(size_t)r * NPAD] = acc[r];
        }
    }
}

// ---------------- edge phase: gather Y (L2-resident), activate, scatter-add ----------------
// Grid-stride, one wave per edge per iteration; lanes stride the 129 channels.
// Software-pipelined: while computing edge e, lanes issue global_prefetch_b8 over
// the next edge's two Y gather rows (9 x 256B lines each) to hide L2 latency.
__global__ void __launch_bounds__(256) k_edge(const int* __restrict__ src,
                                              const int* __restrict__ dst,
                                              const float* __restrict__ ea,
                                              const float* __restrict__ Y,
                                              const float* __restrict__ wfe,   // Wf row 258
                                              const float* __restrict__ wse,   // Ws row 258
                                              const float* __restrict__ bfv,
                                              const float* __restrict__ bsv,
                                              float* __restrict__ xn) {
    const int lane = threadIdx.x & 31;
    const int gw = (blockIdx.x * blockDim.x + threadIdx.x) >> 5;
    const int nw = (gridDim.x * blockDim.x) >> 5;

    for (int e = gw; e < N_EDGES; e += nw) {
        int s = src[e], d = dst[e];
        float a = ea[e];

        // prefetch the next edge's gather rows (2112B per Y row)
        int en = e + nw;
        if (en < N_EDGES) {
            int sn = src[en], dn = dst[en];
            const char* pd = (const char*)(Y + (size_t)dn * NPAD);
            const char* ps = (const char*)(Y + (size_t)sn * NPAD);
            if (lane < 9)       __builtin_prefetch(pd + (size_t)lane * 256, 0, 0);
            else if (lane < 18) __builtin_prefetch(ps + (size_t)(lane - 9) * 256, 0, 0);
        }

        const float* Yd  = Y + (size_t)d * NPAD;
        const float* Ysv = Y + (size_t)s * NPAD;
        float* xd = xn + (size_t)d * CD;
        for (int c = lane; c < CD; c += 32) {
            float mf = Yd[c]        + Ysv[CD + c]   + a * wfe[c] + bfv[c];
            float ms = Yd[2*CD + c] + Ysv[3*CD + c] + a * wse[c] + bsv[c];
            float sig = 1.0f / (1.0f + __expf(-mf));
            float sp  = (ms > 20.0f) ? ms : log1pf(__expf(ms));
            atomicAdd(xd + c, sig * sp);
        }
    }
}

// ---------------- pooling + FC head ----------------

__global__ void k_zero(float* __restrict__ p, int n) {
    int i = blockIdx.x * blockDim.x + threadIdx.x;
    if (i < n) p[i] = 0.0f;
}

__global__ void __launch_bounds__(256) k_pool(const float* __restrict__ x,
                                              const int* __restrict__ batch,
                                              float* __restrict__ gsum,
                                              float* __restrict__ gcnt) {
    int node = (blockIdx.x * blockDim.x + threadIdx.x) >> 5;
    int lane = threadIdx.x & 31;
    if (node >= N_NODES) return;
    int b = batch[node];
    for (int c = lane; c < CD; c += 32)
        atomicAdd(gsum + (size_t)b * CD + c, x[(size_t)node * CD + c]);
    if (lane == 0) atomicAdd(gcnt + b, 1.0f);
}

__global__ void __launch_bounds__(160) k_fc(const float* __restrict__ gsum,
                                            const float* __restrict__ gcnt,
                                            const float* __restrict__ Wfc,
                                            const float* __restrict__ bfc,
                                            const float* __restrict__ Wout,
                                            const float* __restrict__ bout,
                                            float* __restrict__ out) {
    __shared__ float g[CD];
    __shared__ float h[CD];
    int b = blockIdx.x;
    int c = threadIdx.x;
    if (c < CD) g[c] = gsum[(size_t)b * CD + c] / fmaxf(gcnt[b], 1.0f);
    __syncthreads();
    for (int l = 0; l < NFC; ++l) {
        if (c < CD) {
            float acc = bfc[l * CD + c];
            const float* W = Wfc + (size_t)l * CD * CD;
            for (int k = 0; k < CD; ++k) acc += g[k] * W[k * CD + c];
            h[c] = acc;
        }
        __syncthreads();
        if (c < CD) g[c] = h[c];
        __syncthreads();
    }
    if (c == 0) {
        float acc = bout[0];
        for (int k = 0; k < CD; ++k) acc += g[k] * Wout[k];
        out[b] = acc;
    }
}

// ---------------- launcher ----------------

extern "C" void kernel_launch(void* const* d_in, const int* in_sizes, int n_in,
                              void* d_out, int out_size, void* d_ws, size_t ws_size,
                              hipStream_t stream) {
    (void)in_sizes; (void)n_in; (void)out_size; (void)ws_size;
    const int*   atoms = (const int*)  d_in[0];
    const float* pos   = (const float*)d_in[1];
    const int*   ei    = (const int*)  d_in[2];
    const int*   batch = (const int*)  d_in[3];
    const float* emb   = (const float*)d_in[4];
    const float* Wf    = (const float*)d_in[5];
    const float* bf    = (const float*)d_in[6];
    const float* Ws    = (const float*)d_in[7];
    const float* bs    = (const float*)d_in[8];
    const float* Wfc   = (const float*)d_in[9];
    const float* bfc   = (const float*)d_in[10];
    const float* Wout  = (const float*)d_in[11];
    const float* bout  = (const float*)d_in[12];
    const int* src = ei;              // x_j = x[src]
    const int* dst = ei + N_EDGES;    // aggregate at dst

    // workspace carve-up, 256B aligned (~177 MB total)
    char* w = (char*)d_ws;
    auto carve = [&](size_t bytes) { char* p = w; w += (bytes + 255) & ~(size_t)255; return p; };
    float*  xA   = (float*) carve((size_t)N_NODES * CD * 4);
    float*  xB   = (float*) carve((size_t)N_NODES * CD * 4);
    __bf16* xb16 = (__bf16*)carve((size_t)N_NODES * KPAD * 2);
    float*  Y    = (float*) carve((size_t)N_NODES * NPAD * 4);
    __bf16* WT   = (__bf16*)carve((size_t)NPAD * KPAD * 2);
    float*  ea   = (float*) carve((size_t)N_EDGES * 4);
    float*  gsum = (float*) carve((size_t)NUM_GRAPHS * CD * 4);
    float*  gcnt = (float*) carve((size_t)NUM_GRAPHS * 4);

    k_init_x<<<(N_NODES * CD + 255) / 256, 256, 0, stream>>>(atoms, pos, emb, xA);
    k_edge_attr<<<(N_EDGES + 255) / 256, 256, 0, stream>>>(src, dst, pos, ea);

    float* xc = xA;
    float* xn = xB;
    for (int l = 0; l < NLAYERS; ++l) {
        k_cvt_x<<<(N_NODES * KPAD + 255) / 256, 256, 0, stream>>>(xc, xb16);
        k_prep_w<<<(NPAD * KPAD + 255) / 256, 256, 0, stream>>>(
            Wf + (size_t)l * ZD * CD, Ws + (size_t)l * ZD * CD, WT);

        k_node_gemm<<<N_NODES / 16, 256, 0, stream>>>(xb16, WT, Y);

        // residual: x_next starts as x (capture-safe D2D copy)
        hipMemcpyAsync(xn, xc, (size_t)N_NODES * CD * 4, hipMemcpyDeviceToDevice, stream);

        k_edge<<<2048, 256, 0, stream>>>(
            src, dst, ea, Y,
            Wf + (size_t)l * ZD * CD + (size_t)(ZD - 1) * CD,
            Ws + (size_t)l * ZD * CD + (size_t)(ZD - 1) * CD,
            bf + l * CD, bs + l * CD, xn);

        float* t = xc; xc = xn; xn = t;
    }

    k_zero<<<(NUM_GRAPHS * CD + 255) / 256, 256, 0, stream>>>(gsum, NUM_GRAPHS * CD);
    k_zero<<<1, 256, 0, stream>>>(gcnt, NUM_GRAPHS);
    k_pool<<<(size_t)N_NODES * 32 / 256 + 1, 256, 0, stream>>>(xc, batch, gsum, gcnt);
    k_fc<<<NUM_GRAPHS, 160, 0, stream>>>(gsum, gcnt, Wfc, bfc, Wout, bout, (float*)d_out);
}
